// PagedJointAttention_12025908429322
// MI455X (gfx1250) — matrix operand order repaired and checked by reading, not compiled, not run on hardware
//
#include <hip/hip_runtime.h>
#include <math.h>
#include <stdint.h>

typedef __bf16 bf16;
typedef __attribute__((ext_vector_type(16))) __bf16 v16bf;
typedef __attribute__((ext_vector_type(8)))  float  v8f;

#define H_   16
#define D_   64
#define E_   1024
#define S_   2048
#define SC_  6144
#define ST_  8192      /* SC + S */
#define NT_  (ST_/64)  /* 128 tiles of 64 kv rows per head */
#define EPS_ 1e-6f
#define LOG2E_ 1.4426950408889634f

static __device__ __forceinline__ v8f wmma_bf16(v16bf a, v16bf b, v8f c) {
  return __builtin_amdgcn_wmma_f32_16x16x32_bf16(false, a, false, b, (short)0, c,
                                                 false, false);
}

static __device__ __forceinline__ uint32_t pack2_bf16(float a, float b) {
  union { bf16 h[2]; uint32_t u; } v;
  v.h[0] = (bf16)a;
  v.h[1] = (bf16)b;
  return v.u;  // lowers to v_cvt_pk_bf16_f32
}

// A-fragment (16x32 bf16), contiguous k order: k = 16*(lane/16) + s.
static __device__ __forceinline__ v16bf load_frag_a(const bf16* base, int ld) {
  const int lane = threadIdx.x & 31;
  const int m = lane & 15;
  const int g = lane >> 4;
  const bf16* row = base + m * ld;
  v16bf f;
#pragma unroll
  for (int i = 0; i < 8; ++i) {
    int k = g * 16 + 2 * i;
    f[2 * i]     = row[k];
    f[2 * i + 1] = row[k + 1];
  }
  return f;
}

// B-fragment (32x16 bf16): element (k,n) at base[k*strideK + n*strideN].
static __device__ __forceinline__ v16bf load_frag_b(const bf16* base, int strideK,
                                                    int strideN) {
  const int lane = threadIdx.x & 31;
  const int n = lane & 15;
  const int g = lane >> 4;
  const bf16* col = base + n * strideN + g * 16 * strideK;
  v16bf f;
#pragma unroll
  for (int i = 0; i < 8; ++i) {
    f[2 * i]     = col[(2 * i) * strideK];
    f[2 * i + 1] = col[(2 * i + 1) * strideK];
  }
  return f;
}

// CDNA5 async global->LDS copy (GLOBAL_LOAD_ASYNC_TO_LDS_B128, ASYNCcnt).
static __device__ __forceinline__ void async_copy_b128(bf16* lds_dst,
                                                       const bf16* gsrc) {
  uint32_t l = (uint32_t)(uintptr_t)lds_dst;  // low 32 bits = LDS offset
  asm volatile("global_load_async_to_lds_b128 %0, %1, off"
               :
               : "v"(l), "v"(gsrc)
               : "memory");
}
static __device__ __forceinline__ void wait_async_zero() {
  asm volatile("s_wait_asynccnt 0" ::: "memory");
}

// ---------------------------------------------------------------------------
// Kernel 1: stage KV cache -> bf16. K rows RMSNormed, row-major [h][s][d].
// V written tile-transposed [h][tile][d][k64]. Block = (64-row tile, head).
// ---------------------------------------------------------------------------
__global__ __launch_bounds__(256) void stage_kv_kernel(
    const float* __restrict__ k_cache, const float* __restrict__ v_cache,
    const float* __restrict__ norm_k_w, bf16* __restrict__ Kst,
    bf16* __restrict__ VstT) {
  __shared__ bf16 vt[64 * 64];  // [d][k] transposed tile
  const int tile = blockIdx.x;  // 0..SC_/64-1
  const int h = blockIdx.y;
  const int s0 = tile * 64;
  const int tid = threadIdx.x;
  const int wave = tid >> 5;
  const int lane = tid & 31;

  // K: 8 waves x 8 rows each, rmsnorm over D=64 (2 elems/lane)
  const float w0 = norm_k_w[lane * 2], w1 = norm_k_w[lane * 2 + 1];
#pragma unroll
  for (int rr = 0; rr < 8; ++rr) {
    const int s = s0 + wave * 8 + rr;
    const float* r = k_cache + ((long)h * SC_ + s) * D_;
    float x0 = r[lane * 2], x1 = r[lane * 2 + 1];
    float ss = x0 * x0 + x1 * x1;
#pragma unroll
    for (int off = 16; off >= 1; off >>= 1) ss += __shfl_xor(ss, off, 32);
    float inv = rsqrtf(ss * (1.0f / D_) + EPS_);
    bf16* o = Kst + ((long)h * ST_ + s) * D_;
    *(uint32_t*)(o + lane * 2) = pack2_bf16(x0 * inv * w0, x1 * inv * w1);
  }

  // V: coalesced read, transpose through LDS, contiguous coalesced store
#pragma unroll
  for (int i = 0; i < 16; ++i) {
    int idx = tid + i * 256;  // 0..4095
    int r = idx >> 6, c = idx & 63;
    vt[c * 64 + r] = (bf16)v_cache[((long)h * SC_ + s0 + r) * D_ + c];
  }
  __syncthreads();
  bf16* o = VstT + ((long)(h * NT_ + tile)) * 4096;
  const uint4* src = (const uint4*)vt;
  uint4* dst = (uint4*)o;
#pragma unroll
  for (int i = 0; i < 2; ++i) dst[tid + i * 256] = src[tid + i * 256];
}

// ---------------------------------------------------------------------------
// Kernel 2: QKV projection GEMM (C = x @ W^T + b), WMMA bf16, 64x64 C tile,
// K-step 64. Epilogue: bias + per-head RMSNorm (Q,K; 1/sqrt(D) folded into
// Q); V stored tile-transposed. Packed bf16 conversions throughout.
// ---------------------------------------------------------------------------
__global__ __launch_bounds__(256) void qkv_proj_kernel(
    const float* __restrict__ x, const float* __restrict__ Wq,
    const float* __restrict__ bq, const float* __restrict__ Wk,
    const float* __restrict__ bk, const float* __restrict__ Wv,
    const float* __restrict__ bv, const float* __restrict__ nqw,
    const float* __restrict__ nkw, bf16* __restrict__ Qst,
    bf16* __restrict__ Kst, bf16* __restrict__ VstT) {
  __shared__ bf16 xt[64 * 64];
  __shared__ bf16 wt[64 * 64];
  __shared__ float ct[64 * 64];

  const int kind = blockIdx.z;  // 0=Q 1=K 2=V
  const float* W    = (kind == 0) ? Wq : (kind == 1) ? Wk : Wv;
  const float* bias = (kind == 0) ? bq : (kind == 1) ? bk : bv;
  const int t0 = blockIdx.x * 64;
  const int e0 = blockIdx.y * 64;  // head-aligned
  const int tid = threadIdx.x;
  const int wave = tid >> 5;
  const int waveM = wave >> 1;
  const int waveN = wave & 1;

  v8f acc0 = {}, acc1 = {};
  for (int kk = 0; kk < E_; kk += 64) {
    __syncthreads();
#pragma unroll
    for (int i = 0; i < 8; ++i) {
      int idx = (tid + i * 256) * 2;  // even element over 64x64 tile
      int r = idx >> 6, c = idx & 63;
      const float2 xv = *(const float2*)(x + (long)(t0 + r) * E_ + kk + c);
      const float2 wv = *(const float2*)(W + (long)(e0 + r) * E_ + kk + c);
      *(uint32_t*)(xt + idx) = pack2_bf16(xv.x, xv.y);
      *(uint32_t*)(wt + idx) = pack2_bf16(wv.x, wv.y);
    }
    __syncthreads();
    v16bf a0 = load_frag_a(xt + waveM * 16 * 64, 64);
    v16bf a1 = load_frag_a(xt + waveM * 16 * 64 + 32, 64);
    acc0 = wmma_bf16(a0, load_frag_b(wt + (waveN * 32 + 0) * 64, 1, 64), acc0);
    acc0 = wmma_bf16(a1, load_frag_b(wt + (waveN * 32 + 0) * 64 + 32, 1, 64), acc0);
    acc1 = wmma_bf16(a0, load_frag_b(wt + (waveN * 32 + 16) * 64, 1, 64), acc1);
    acc1 = wmma_bf16(a1, load_frag_b(wt + (waveN * 32 + 16) * 64 + 32, 1, 64), acc1);
  }
  __syncthreads();
  {  // C fragments -> LDS for the row/col-wise epilogue
    const int lane = tid & 31;
    const int n = lane & 15, hh = lane >> 4;
#pragma unroll
    for (int r = 0; r < 8; ++r) {
      int m = waveM * 16 + r + 8 * hh;
      ct[m * 64 + waveN * 32 + n]      = acc0[r];
      ct[m * 64 + waveN * 32 + 16 + n] = acc1[r];
    }
  }
  __syncthreads();
  const int h = e0 >> 6;
  if (kind == 2) {
    // V: bias + write transposed tile [d][tloc], packed stores
    const int col = tid >> 2;  // d
    const int sub = tid & 3;
    bf16* o = VstT + ((long)(h * NT_ + (SC_ + t0) / 64)) * 4096 + col * 64 + sub * 16;
    const float b = bias[e0 + col];
#pragma unroll
    for (int i = 0; i < 8; ++i) {
      float a0 = ct[(sub * 16 + 2 * i) * 64 + col] + b;
      float a1 = ct[(sub * 16 + 2 * i + 1) * 64 + col] + b;
      *(uint32_t*)(o + 2 * i) = pack2_bf16(a0, a1);
    }
    return;
  }
  // Q/K: bias + rmsnorm per row (4 lanes/row, 16 cols each)
  const int row = tid >> 2;
  const int sub = tid & 3;
  float v[16];
  float ssq = 0.f;
#pragma unroll
  for (int i = 0; i < 16; ++i) {
    int c = sub * 16 + i;
    float val = ct[row * 64 + c] + bias[e0 + c];
    v[i] = val;
    ssq += val * val;
  }
  ssq += __shfl_xor(ssq, 1, 32);
  ssq += __shfl_xor(ssq, 2, 32);
  float inv = rsqrtf(ssq * (1.0f / D_) + EPS_);
  const long t = t0 + row;
  if (kind == 0) {
    const float qs = 0.125f;  // 1/sqrt(64)
    bf16* o = Qst + ((long)h * S_ + t) * D_ + sub * 16;
#pragma unroll
    for (int i = 0; i < 8; ++i)
      *(uint32_t*)(o + 2 * i) =
          pack2_bf16(v[2 * i] * inv * nqw[sub * 16 + 2 * i] * qs,
                     v[2 * i + 1] * inv * nqw[sub * 16 + 2 * i + 1] * qs);
  } else {
    bf16* o = Kst + ((long)h * ST_ + (SC_ + t)) * D_ + sub * 16;
#pragma unroll
    for (int i = 0; i < 8; ++i)
      *(uint32_t*)(o + 2 * i) =
          pack2_bf16(v[2 * i] * inv * nkw[sub * 16 + 2 * i],
                     v[2 * i + 1] * inv * nkw[sub * 16 + 2 * i + 1]);
  }
}

// ---------------------------------------------------------------------------
// Kernel 3: flash attention with double-buffered async global->LDS tile
// streaming. Block = (64 q rows, head), 4 waves x 16 q rows.
// ---------------------------------------------------------------------------
static __device__ __forceinline__ void load_tiles_async(const bf16* gk,
                                                        const bf16* gv,
                                                        bf16* ldsK, bf16* ldsV,
                                                        int tid) {
#pragma unroll
  for (int i = 0; i < 4; ++i) {
    int e = (tid + i * 128) * 8;  // 8 bf16 = 16B per op
    async_copy_b128(ldsK + e, gk + e);
    async_copy_b128(ldsV + e, gv + e);
  }
}

__global__ __launch_bounds__(128) void flash_attn_kernel(
    const bf16* __restrict__ Qst, const bf16* __restrict__ Kst,
    const bf16* __restrict__ VstT, bf16* __restrict__ Ost) {
  __shared__ bf16 Kt[2][64 * 64];  // [k][d] row-major
  __shared__ bf16 Vt[2][64 * 64];  // [d][k] transposed
  __shared__ bf16 Pt[4][16 * 64];

  const int h = blockIdx.y;
  const int q0 = blockIdx.x * 64;
  const int tid = threadIdx.x;
  const int wave = tid >> 5;
  const int lane = tid & 31;
  const int qr = q0 + wave * 16;

  const bf16* Qbase = Qst + ((long)h * S_ + qr) * D_;
  v16bf aq0 = load_frag_a(Qbase, D_);
  v16bf aq1 = load_frag_a(Qbase + 32, D_);

  v8f accO[4] = {};
  float rowm[8], rowl[8];
#pragma unroll
  for (int r = 0; r < 8; ++r) { rowm[r] = -3.0e38f; rowl[r] = 0.f; }

  const bf16* Kh = Kst + (long)h * ST_ * D_;
  const bf16* VhT = VstT + (long)h * NT_ * 4096;

  load_tiles_async(Kh, VhT, Kt[0], Vt[0], tid);
  wait_async_zero();
  __syncthreads();
  int cur = 0;

  for (int kt = 0; kt < ST_; kt += 64) {
    if (kt + 64 < ST_)  // prefetch next tile into the other buffer
      load_tiles_async(Kh + (long)(kt + 64) * D_, VhT + (long)((kt >> 6) + 1) * 4096,
                       Kt[cur ^ 1], Vt[cur ^ 1], tid);

    // S = Q * K^T : 16x64 scores per wave
    v8f s[4];
#pragma unroll
    for (int nb = 0; nb < 4; ++nb) {
      v8f acc = {};
      acc = wmma_bf16(aq0, load_frag_b(Kt[cur] + nb * 16 * D_, 1, D_), acc);
      acc = wmma_bf16(aq1, load_frag_b(Kt[cur] + nb * 16 * D_ + 32, 1, D_), acc);
      s[nb] = acc;
    }

    // online softmax
    float pexp[4][8];
#pragma unroll
    for (int r = 0; r < 8; ++r) {
      float cm = s[0][r];
#pragma unroll
      for (int nb = 1; nb < 4; ++nb) cm = fmaxf(cm, s[nb][r]);
#pragma unroll
      for (int off = 8; off >= 1; off >>= 1) cm = fmaxf(cm, __shfl_xor(cm, off, 32));
      float nm = fmaxf(rowm[r], cm);
      float corr = exp2f((rowm[r] - nm) * LOG2E_);
      rowm[r] = nm;
      float rs = 0.f;
#pragma unroll
      for (int nb = 0; nb < 4; ++nb) {
        float p = exp2f((s[nb][r] - nm) * LOG2E_);
        pexp[nb][r] = p;
        rs += p;
      }
#pragma unroll
      for (int off = 8; off >= 1; off >>= 1) rs += __shfl_xor(rs, off, 32);
      rowl[r] = rowl[r] * corr + rs;
#pragma unroll
      for (int nb = 0; nb < 4; ++nb) accO[nb][r] *= corr;
    }

    // P (C-layout) -> per-wave LDS region -> A-layout. Same-wave LDS ops are
    // processed in order (DScnt in-order); threadfence orders codegen only.
    {
      const int n = lane & 15, hh = lane >> 4;
#pragma unroll
      for (int r = 0; r < 8; ++r) {
        int m = r + 8 * hh;
#pragma unroll
        for (int nb = 0; nb < 4; ++nb)
          Pt[wave][m * 64 + nb * 16 + n] = (bf16)pexp[nb][r];
      }
    }
    __threadfence_block();
    v16bf ap0 = load_frag_a(Pt[wave], 64);
    v16bf ap1 = load_frag_a(Pt[wave] + 32, 64);

    // O += P * V, V transposed in LDS -> contiguous B-fragment pairs
#pragma unroll
    for (int nb = 0; nb < 4; ++nb) {
      accO[nb] = wmma_bf16(ap0, load_frag_b(Vt[cur] + (nb * 16) * 64, 1, 64), accO[nb]);
      accO[nb] = wmma_bf16(ap1, load_frag_b(Vt[cur] + (nb * 16) * 64 + 32, 1, 64), accO[nb]);
    }

    wait_async_zero();  // next tile resident
    __syncthreads();
    cur ^= 1;
  }

  // epilogue: normalize, write bf16 [t][h*64+d]
  const int n = lane & 15, hh = lane >> 4;
#pragma unroll
  for (int r = 0; r < 8; ++r) {
    int m = r + 8 * hh;
    float invl = 1.0f / rowl[r];
#pragma unroll
    for (int nb = 0; nb < 4; ++nb)
      Ost[(long)(qr + m) * E_ + h * D_ + nb * 16 + n] =
          (bf16)(accO[nb][r] * invl);
  }
}

// ---------------------------------------------------------------------------
// Kernel 4: output projection (out = O @ Wo^T + bo), f32 result, K-step 64.
// ---------------------------------------------------------------------------
__global__ __launch_bounds__(256) void out_proj_kernel(
    const bf16* __restrict__ Ost, const float* __restrict__ Wo,
    const float* __restrict__ bo, float* __restrict__ out) {
  __shared__ bf16 at[64 * 64];
  __shared__ bf16 wt[64 * 64];
  const int t0 = blockIdx.x * 64;
  const int e0 = blockIdx.y * 64;
  const int tid = threadIdx.x;
  const int wave = tid >> 5;
  const int waveM = wave >> 1;
  const int waveN = wave & 1;
  v8f acc0 = {}, acc1 = {};
  for (int kk = 0; kk < E_; kk += 64) {
    __syncthreads();
#pragma unroll
    for (int i = 0; i < 8; ++i) {
      int idx = (tid + i * 256) * 2;
      int r = idx >> 6, c = idx & 63;
      *(uint32_t*)(at + idx) = *(const uint32_t*)(Ost + (long)(t0 + r) * E_ + kk + c);
      const float2 wv = *(const float2*)(Wo + (long)(e0 + r) * E_ + kk + c);
      *(uint32_t*)(wt + idx) = pack2_bf16(wv.x, wv.y);
    }
    __syncthreads();
    v16bf a0 = load_frag_a(at + waveM * 16 * 64, 64);
    v16bf a1 = load_frag_a(at + waveM * 16 * 64 + 32, 64);
    acc0 = wmma_bf16(a0, load_frag_b(wt + (waveN * 32 + 0) * 64, 1, 64), acc0);
    acc0 = wmma_bf16(a1, load_frag_b(wt + (waveN * 32 + 0) * 64 + 32, 1, 64), acc0);
    acc1 = wmma_bf16(a0, load_frag_b(wt + (waveN * 32 + 16) * 64, 1, 64), acc1);
    acc1 = wmma_bf16(a1, load_frag_b(wt + (waveN * 32 + 16) * 64 + 32, 1, 64), acc1);
  }
  const int lane = tid & 31;
  const int n = lane & 15, hh = lane >> 4;
#pragma unroll
  for (int r = 0; r < 8; ++r) {
    int m = r + 8 * hh;
    long row = t0 + waveM * 16 + m;
    int c0 = e0 + waveN * 32 + n;
    out[row * E_ + c0]      = acc0[r] + bo[c0];
    out[row * E_ + c0 + 16] = acc1[r] + bo[c0 + 16];
  }
}

// ---------------------------------------------------------------------------
extern "C" void kernel_launch(void* const* d_in, const int* in_sizes, int n_in,
                              void* d_out, int out_size, void* d_ws,
                              size_t ws_size, hipStream_t stream) {
  const float* x       = (const float*)d_in[0];
  const float* k_cache = (const float*)d_in[1];
  const float* v_cache = (const float*)d_in[2];
  const float* Wq      = (const float*)d_in[3];
  const float* bq      = (const float*)d_in[4];
  const float* Wk      = (const float*)d_in[5];
  const float* bk      = (const float*)d_in[6];
  const float* Wv      = (const float*)d_in[7];
  const float* bv      = (const float*)d_in[8];
  const float* Wo      = (const float*)d_in[9];
  const float* bo      = (const float*)d_in[10];
  const float* nqw     = (const float*)d_in[11];
  const float* nkw     = (const float*)d_in[12];
  float* out = (float*)d_out;

  char* ws = (char*)d_ws;
  bf16* Qst  = (bf16*)ws; ws += (size_t)H_ * S_  * D_ * sizeof(bf16);  // 4 MB
  bf16* Kst  = (bf16*)ws; ws += (size_t)H_ * ST_ * D_ * sizeof(bf16);  // 16 MB
  bf16* VstT = (bf16*)ws; ws += (size_t)H_ * ST_ * D_ * sizeof(bf16);  // 16 MB
  bf16* Ost  = (bf16*)ws;                                              // 4 MB

  dim3 g1(SC_ / 64, H_);
  stage_kv_kernel<<<g1, 256, 0, stream>>>(k_cache, v_cache, nkw, Kst, VstT);

  dim3 g2(S_ / 64, E_ / 64, 3);
  qkv_proj_kernel<<<g2, 256, 0, stream>>>(x, Wq, bq, Wk, bk, Wv, bv, nqw, nkw,
                                          Qst, Kst, VstT);

  dim3 g3(S_ / 64, H_);
  flash_attn_kernel<<<g3, 128, 0, stream>>>(Qst, Kst, VstT, Ost);

  dim3 g4(S_ / 64, E_ / 64);
  out_proj_kernel<<<g4, 256, 0, stream>>>(Ost, Wo, bo, out);
}
